// CRFLoss_13915694039461
// MI455X (gfx1250) — matrix operand order repaired and checked by reading, not compiled, not run on hardware
//
#include <hip/hip_runtime.h>
#include <hip/hip_bf16.h>

// CRF forward score via exp-domain associative block products + f16 WMMA.
// K=512 states, T=32768 steps, NB=64 chains of BSTEPS=512, 4 column-slices/chain.

#define KK 512
#define TT 32768
#define NB 64
#define BSTEPS 512      // NB * BSTEPS == TT
#define SCOLS 128       // columns per slice (KK / 4)
#define SPAD 520        // padded LDS column stride (halves) to spread banks

typedef __attribute__((ext_vector_type(16))) _Float16 v16h;
typedef __attribute__((ext_vector_type(8)))  _Float16 v8h;
typedef __attribute__((ext_vector_type(8)))  float    v8f;

static __device__ __forceinline__ v16h pack16(const _Float16* lo, const _Float16* hi) {
  v8h l = *(const v8h*)lo;
  v8h h = *(const v8h*)hi;
  v16h r;
#pragma unroll
  for (int x = 0; x < 8; ++x) { r[x] = l[x]; r[x + 8] = h[x]; }
  return r;
}

// ---------------- prep kernels ----------------
__global__ void prep_E(const float* __restrict__ tr, _Float16* __restrict__ E) {
  int i = blockIdx.x * 256 + threadIdx.x;           // KK*KK = 262144 elems
  if (i < KK * KK) E[i] = (_Float16)expf(tr[i]);
}

__global__ void prep_obs(const float* __restrict__ obs, float* __restrict__ obsE) {
  size_t idx = (size_t)blockIdx.x * 256 + threadIdx.x;   // TT*KK elems
  if (idx < (size_t)TT * KK) {
    int t = (int)(idx >> 9);
    int i = (int)(idx & 511);
    obsE[idx] = expf(obs[(size_t)i * TT + t]);           // transpose + exp
  }
}

// ---------------- phase 1: block operators G_b ----------------
// grid = NB*4 blocks, 512 threads (16 waves). Each block owns one (chain, col-slice).
// Wave tiling: 8 Mblocks x 2 Nblocks; each wave: 4 Mtiles x 4 Ntiles (16 WMMA tiles).
__global__ __launch_bounds__(512)
void phase1(const _Float16* __restrict__ Eg, const float* __restrict__ obsE,
            _Float16* __restrict__ G, float* __restrict__ Ls) {
  __shared__ __align__(16) _Float16 Sh[SCOLS][SPAD];  // running product slice, col-major
  __shared__ float  colsum[SCOLS];
  __shared__ double lsc[SCOLS];

  const int bid = blockIdx.x;
  const int chain = bid >> 2;
  const int slice = bid & 3;
  const int colbase = slice * SCOLS;

  const int tid  = threadIdx.x;
  const int lane = tid & 31;
  const int wave = tid >> 5;
  const int l16  = lane & 15;
  const int lhi  = lane >> 4;           // 0 or 1 (wave32 half)
  const int Mblock = wave & 7;          // 0..7 -> Mtiles 4*Mblock..+3
  const int Nblock = wave >> 3;         // 0..1 -> Ntiles 4*Nblock..+3

  // init slice = identity columns of this slice; scales = 0
  for (int idx = tid; idx < SCOLS * KK; idx += 512) {
    int c = idx >> 9, k = idx & 511;
    Sh[c][k] = (_Float16)((k == colbase + c) ? 1.0f : 0.0f);
  }
  if (tid < SCOLS) { colsum[tid] = 0.0f; lsc[tid] = 0.0; }
  __syncthreads();

  const int t0 = chain * BSTEPS;
  for (int s = 0; s < BSTEPS; ++s) {
    const int t = t0 + s;

    // ---- acc = E * S  (f16 WMMA, f32 accumulate) ----
    v8f acc[4][4];
#pragma unroll
    for (int mi = 0; mi < 4; ++mi)
#pragma unroll
      for (int ni = 0; ni < 4; ++ni)
#pragma unroll
        for (int r = 0; r < 8; ++r) acc[mi][ni][r] = 0.0f;

    for (int kc = 0; kc < 16; ++kc) {
      const int kb = kc * 32;
      v16h a[4];
#pragma unroll
      for (int mi = 0; mi < 4; ++mi) {
        const int row = (Mblock * 4 + mi) * 16 + l16;
        const _Float16* p = Eg + (size_t)row * KK + kb + lhi * 8;
        a[mi] = pack16(p, p + 16);      // A: K halves {0..7}/{16..23} (+8 for lanes>=16)
      }
#pragma unroll
      for (int ni = 0; ni < 4; ++ni) {
        const int col = (Nblock * 4 + ni) * 16 + l16;
        const _Float16* q = &Sh[col][kb + lhi * 8];
        v16h b = pack16(q, q + 16);     // B: K halves {0..7}/{16..23} (+8 for lanes>=16)
#pragma unroll
        for (int mi = 0; mi < 4; ++mi)
          acc[mi][ni] = __builtin_amdgcn_wmma_f32_16x16x32_f16(
              false, a[mi], false, b, (short)0, acc[mi][ni], false, false);
      }
    }
    __syncthreads();   // all reads of Sh done; colsum is zeroed

    // ---- scale rows by exp(o_t), accumulate column sums ----
    const float* ob = obsE + (size_t)t * KK;
    float part[4];
#pragma unroll
    for (int ni = 0; ni < 4; ++ni) part[ni] = 0.0f;
#pragma unroll
    for (int mi = 0; mi < 4; ++mi) {
      const int row0 = (Mblock * 4 + mi) * 16 + lhi * 8;
      float e[8];
#pragma unroll
      for (int r = 0; r < 8; ++r) e[r] = ob[row0 + r];
#pragma unroll
      for (int ni = 0; ni < 4; ++ni)
#pragma unroll
        for (int r = 0; r < 8; ++r) {
          acc[mi][ni][r] *= e[r];
          part[ni] += acc[mi][ni][r];
        }
    }
#pragma unroll
    for (int ni = 0; ni < 4; ++ni) part[ni] += __shfl_xor(part[ni], 16, 32);
    if (lhi == 0) {
#pragma unroll
      for (int ni = 0; ni < 4; ++ni)
        atomicAdd(&colsum[(Nblock * 4 + ni) * 16 + l16], part[ni]);
    }
    __syncthreads();

    // ---- normalize columns (sum=1), write back f16, update log-scales ----
#pragma unroll
    for (int ni = 0; ni < 4; ++ni) {
      const int c = (Nblock * 4 + ni) * 16 + l16;
      const float cs = colsum[c];
      const float inv = (cs > 0.0f) ? (1.0f / cs) : 0.0f;
#pragma unroll
      for (int mi = 0; mi < 4; ++mi) {
        const int row0 = (Mblock * 4 + mi) * 16 + lhi * 8;
        v8h vh;
#pragma unroll
        for (int r = 0; r < 8; ++r) vh[r] = (_Float16)(acc[mi][ni][r] * inv);
        *(v8h*)&Sh[c][row0] = vh;
      }
    }
    if (Mblock == 0 && lhi == 0) {
#pragma unroll
      for (int ni = 0; ni < 4; ++ni) {
        const int c = (Nblock * 4 + ni) * 16 + l16;
        const float cs = colsum[c];
        lsc[c] += (double)logf(cs > 0.0f ? cs : 1e-30f);
      }
    }
    __syncthreads();
    if (tid < SCOLS) colsum[tid] = 0.0f;   // safe: next read is after post-matmul barrier
  }

  // ---- store G_b slice (f16 col-major) and scales ----
  for (int idx = tid; idx < SCOLS * KK; idx += 512) {
    int c = idx >> 9, k = idx & 511;
    G[(size_t)chain * KK * KK + (size_t)(colbase + c) * KK + k] = Sh[c][k];
  }
  if (tid < SCOLS) Ls[chain * KK + colbase + tid] = (float)lsc[tid];
}

// ---------------- gold path score ----------------
__global__ void gold_zero(float* g) { if (threadIdx.x == 0 && blockIdx.x == 0) *g = 0.0f; }

__global__ void gold_kernel(const float* __restrict__ tr, const float* __restrict__ obs,
                            const int* __restrict__ tags, float* __restrict__ g) {
  __shared__ float red[256];
  float s = 0.0f;
  for (int t = blockIdx.x * 256 + threadIdx.x; t < TT - 1; t += gridDim.x * 256) {
    const int cur = tags[t], nxt = tags[t + 1];
    s += tr[(size_t)nxt * KK + cur] + obs[(size_t)nxt * TT + t];  // obs.T[t, nxt]
  }
  red[threadIdx.x] = s;
  __syncthreads();
  for (int off = 128; off > 0; off >>= 1) {
    if (threadIdx.x < off) red[threadIdx.x] += red[threadIdx.x + off];
    __syncthreads();
  }
  if (threadIdx.x == 0) atomicAdd(g, red[0]);
}

// ---------------- phase 2: sequential log-matvec combine ----------------
__global__ __launch_bounds__(512)
void phase2(const _Float16* __restrict__ G, const float* __restrict__ Ls,
            const float* __restrict__ gold, float* __restrict__ out) {
  __shared__ float red[512];
  __shared__ float w[512];
  const int i = threadIdx.x;
  float fv = 0.0f;                       // fv_0 = 0 (w_0 = ones)
  for (int b = 0; b < NB; ++b) {
    const float v = fv + Ls[b * KK + i];
    red[i] = v;
    for (int off = 256; off > 0; off >>= 1) {
      __syncthreads();
      if (i < off) red[i] = fmaxf(red[i], red[i + off]);
    }
    __syncthreads();
    const float m = red[0];
    w[i] = expf(v - m);
    __syncthreads();
    const _Float16* Gb = G + (size_t)b * KK * KK;
    float ssum = 0.0f;
    for (int j = 0; j < KK; ++j)
      ssum += (float)Gb[(size_t)j * KK + i] * w[j];   // coalesced over i
    fv = logf(fmaxf(ssum, 1e-37f)) + m;
    __syncthreads();
  }
  // final logsumexp over fv
  red[i] = fv;
  for (int off = 256; off > 0; off >>= 1) {
    __syncthreads();
    if (i < off) red[i] = fmaxf(red[i], red[i + off]);
  }
  __syncthreads();
  const float m2 = red[0];
  __syncthreads();
  red[i] = expf(fv - m2);
  for (int off = 256; off > 0; off >>= 1) {
    __syncthreads();
    if (i < off) red[i] += red[i + off];
  }
  __syncthreads();
  if (i == 0) out[0] = (logf(red[0]) + m2) - gold[0];
}

// ---------------- host launcher ----------------
extern "C" void kernel_launch(void* const* d_in, const int* in_sizes, int n_in,
                              void* d_out, int out_size, void* d_ws, size_t ws_size,
                              hipStream_t stream) {
  (void)in_sizes; (void)n_in; (void)out_size; (void)ws_size;
  const float* observes    = (const float*)d_in[0];   // (K,T) f32
  const int*   tags        = (const int*)d_in[1];     // (T,) i32
  const float* transitions = (const float*)d_in[2];   // (K,K) f32
  // d_in[3] = length scalar (== TT), compile-time constant here.

  char* ws = (char*)d_ws;
  const size_t SZ_OBSE = (size_t)TT * KK * 4;         // 64 MB
  const size_t SZ_E16  = (size_t)KK * KK * 2;         // 512 KB
  const size_t SZ_G    = (size_t)NB * KK * KK * 2;    // 32 MB
  const size_t SZ_LS   = (size_t)NB * KK * 4;         // 128 KB
  float*    obsE = (float*)ws;
  _Float16* E16  = (_Float16*)(ws + SZ_OBSE);
  _Float16* G    = (_Float16*)(ws + SZ_OBSE + SZ_E16);
  float*    Ls   = (float*)(ws + SZ_OBSE + SZ_E16 + SZ_G);
  float*    gacc = (float*)(ws + SZ_OBSE + SZ_E16 + SZ_G + SZ_LS);

  prep_E  <<<(KK * KK + 255) / 256, 256, 0, stream>>>(transitions, E16);
  prep_obs<<<(int)(((size_t)TT * KK + 255) / 256), 256, 0, stream>>>(observes, obsE);
  phase1  <<<NB * 4, 512, 0, stream>>>(E16, obsE, G, Ls);
  gold_zero<<<1, 64, 0, stream>>>(gacc);
  gold_kernel<<<128, 256, 0, stream>>>(transitions, observes, tags, gacc);
  phase2  <<<1, 512, 0, stream>>>(G, Ls, gacc, (float*)d_out);
}
